// Generator_30339648979556
// MI455X (gfx1250) — matrix operand order repaired and checked by reading, not compiled, not run on hardware
//
#include <hip/hip_runtime.h>
#include <hip/hip_bf16.h>
#include <math.h>

typedef __attribute__((ext_vector_type(16))) __bf16          v16bf;
typedef __attribute__((ext_vector_type(8)))  unsigned int    v8u;
typedef __attribute__((ext_vector_type(8)))  float           v8f;
typedef int v4i __attribute__((__vector_size__(16)));
typedef unsigned int u32x4 __attribute__((__vector_size__(16)));
typedef int i32x4 __attribute__((__vector_size__(16)));
typedef int i32x8 __attribute__((__vector_size__(32)));

#define SLOPEV 0.2f

#if defined(__HIP_DEVICE_COMPILE__) && \
    __has_builtin(__builtin_amdgcn_global_load_async_to_lds_b128) && \
    __has_builtin(__builtin_amdgcn_s_wait_asynccnt)
#define HAVE_ASYNC 1
#else
#define HAVE_ASYNC 0
#endif

#if defined(__HIP_DEVICE_COMPILE__) && \
    __has_builtin(__builtin_amdgcn_tensor_load_to_lds) && \
    __has_builtin(__builtin_amdgcn_s_wait_tensorcnt)
#define HAVE_TDM 1
#else
#define HAVE_TDM 0
#endif

#define TO_GLOBAL(p) ((__attribute__((address_space(1))) v4i*)(unsigned long long)(const void*)(p))
#define TO_LDS(p)    ((__attribute__((address_space(3))) v4i*)(unsigned int)(unsigned long long)(const void*)(p))

static __device__ inline unsigned short f2bf(float f) {
  unsigned int u = __float_as_uint(f);
  u += 0x7FFFu + ((u >> 16) & 1u);
  return (unsigned short)(u >> 16);
}
static __device__ inline float bflo(unsigned int v) { return __uint_as_float(v << 16); }
static __device__ inline float bfhi(unsigned int v) { return __uint_as_float(v & 0xFFFF0000u); }
static __device__ inline float lrelu(float v) { return v > 0.f ? v : SLOPEV * v; }

static __device__ inline v8f wmma_bf16(v16bf a, v16bf b, v8f c) {
  return __builtin_amdgcn_wmma_f32_16x16x32_bf16(false, a, false, b, (short)0, c,
                                                 false, false);
}

static __device__ inline v16bf pack2(uint4 a, uint4 b) {
  v8u u;
  u[0] = a.x; u[1] = a.y; u[2] = a.z; u[3] = a.w;
  u[4] = b.x; u[5] = b.y; u[6] = b.z; u[7] = b.w;
  return __builtin_bit_cast(v16bf, u);
}

// ---- WMMA fragment loaders (contiguous k order, wave32) ----
// A (16x32): lane<16 -> K [0..15]; lane>=16 -> K [16..31]
static __device__ inline v16bf loadA_vec(const unsigned short* rowptr) {
  int lane = threadIdx.x & 31;
  int kb = (lane & 16) ? 16 : 0;
  uint4 a = *(const uint4*)(rowptr + kb);
  uint4 b = *(const uint4*)(rowptr + kb + 8);
  return pack2(a, b);
}
// B (32x16): lane holds column n=lane&15; K = [0..15] (lane<16) or [16..31]
static __device__ inline v16bf loadB_vec(const unsigned short* colptr, bool pred) {
  int lane = threadIdx.x & 31;
  int kb = (lane & 16) ? 16 : 0;
  uint4 a = {0u, 0u, 0u, 0u}, b = {0u, 0u, 0u, 0u};
  if (pred) {
    a = *(const uint4*)(colptr + kb);
    b = *(const uint4*)(colptr + kb + 8);
  }
  return pack2(a, b);
}
static __device__ inline v16bf loadB_vec_np(const unsigned short* colptr) {
  int lane = threadIdx.x & 31;
  int kb = (lane & 16) ? 16 : 0;
  uint4 a = *(const uint4*)(colptr + kb);
  uint4 b = *(const uint4*)(colptr + kb + 8);
  return pack2(a, b);
}

#if HAVE_TDM
// 1D bf16 tile copy via Tensor Data Mover (ISA ch.8 D#: 1-row tile, data_size=2B)
static __device__ inline void tdm_load_1d(const void* gsrc, void* ldst, unsigned nelem) {
  unsigned long long ga = (unsigned long long)gsrc;
  unsigned lds = (unsigned)(unsigned long long)ldst;  // generic low 32 = LDS offset
  u32x4 g0;
  g0[0] = 1u;                                    // count=1, user descriptor
  g0[1] = lds;                                   // lds_addr
  g0[2] = (unsigned)ga;                          // global_addr[31:0]
  g0[3] = ((unsigned)(ga >> 32) & 0x01FFFFFFu) | (2u << 30);  // addr[56:32], type=2
  i32x8 g1;
  g1[0] = (int)(1u << 16);                       // data_size=1 -> 2 bytes
  g1[1] = (int)((nelem & 0xFFFFu) << 16);        // tensor_dim0[15:0]
  g1[2] = (int)(((nelem >> 16) & 0xFFFFu) | (1u << 16));  // tensor_dim0 hi, tensor_dim1=1
  g1[3] = (int)((nelem & 0xFFFFu) << 16);        // tensor_dim1 hi=0, tile_dim0=nelem
  g1[4] = 0;                                     // tile_dim1=0, tile_dim2=0
  g1[5] = (int)nelem;                            // tensor_dim0_stride lo
  g1[6] = 0;
  g1[7] = 0;
  i32x4 z = {0, 0, 0, 0};
#if defined(__clang_major__) && __clang_major__ >= 23
  i32x8 z8 = {0, 0, 0, 0, 0, 0, 0, 0};
  __builtin_amdgcn_tensor_load_to_lds(g0, g1, z, z, z8, 0);
#else
  __builtin_amdgcn_tensor_load_to_lds(g0, g1, z, z, 0);
#endif
}
#endif

// hash-based standard normal (stand-in for jax.random.normal)
static __device__ inline float gnoise(unsigned int b, unsigned int h, unsigned int w,
                                      unsigned int c) {
  unsigned int s = b * 2654435761u ^ (h * 40503u + 12345u) ^ (w * 69069u) ^ (c * 362437u);
  s ^= s >> 16; s *= 0x7feb352du; s ^= s >> 15; s *= 0x846ca68bu; s ^= s >> 16;
  unsigned int t = s * 0x9E3779B9u + 0x85EBCA6Bu;
  t ^= t >> 16; t *= 0x7feb352du; t ^= t >> 15;
  float u1 = ((float)s + 1.0f) * (1.0f / 4294967296.0f);
  float u2 = (float)t * (1.0f / 4294967296.0f);
  return sqrtf(-2.0f * __logf(u1 + 1e-12f)) * __cosf(6.2831853f * u2);
}

// ================= kernels =================

// f32 -> bf16 bulk convert (8 elems/thread, b128 stores)
__global__ void k_f2bf(const float* __restrict__ src, unsigned short* __restrict__ dst,
                       int total8) {
  int idx = blockIdx.x * blockDim.x + threadIdx.x;
  if (idx >= total8) return;
  float4 a = ((const float4*)src)[2 * idx];
  float4 b = ((const float4*)src)[2 * idx + 1];
  uint4 o;
  o.x = (unsigned)f2bf(a.x) | ((unsigned)f2bf(a.y) << 16);
  o.y = (unsigned)f2bf(a.z) | ((unsigned)f2bf(a.w) << 16);
  o.z = (unsigned)f2bf(b.x) | ((unsigned)f2bf(b.y) << 16);
  o.w = (unsigned)f2bf(b.z) | ((unsigned)f2bf(b.w) << 16);
  ((uint4*)dst)[idx] = o;
}

// Mapping layer on bf16 operands: out = lrelu(x @ W^T + b); writes f32 + bf16
__global__ __launch_bounds__(32) void k_map2(const unsigned short* __restrict__ xb,
                                             const unsigned short* __restrict__ Wb,
                                             const float* __restrict__ bias,
                                             float* __restrict__ outf,
                                             unsigned short* __restrict__ outb) {
  int nt = blockIdx.x;
  int lane = threadIdx.x & 31;
  int li = lane & 15;
  v8f acc;
#pragma unroll
  for (int r = 0; r < 8; ++r) acc[r] = 0.f;
  const unsigned short* arow = xb + (size_t)li * 512;
  const unsigned short* brow = Wb + (size_t)(nt * 16 + li) * 512;
#pragma unroll
  for (int k0 = 0; k0 < 512; k0 += 32) {
    v16bf A = loadA_vec(arow + k0);
    v16bf B = loadB_vec_np(brow + k0);
    acc = wmma_bf16(A, B, acc);
  }
  int n = nt * 16 + li;
  int mb = (lane & 16) ? 8 : 0;
  float bn = bias[n];
#pragma unroll
  for (int r = 0; r < 8; ++r) {
    float v = lrelu(acc[r] + bn);
    outf[(size_t)(mb + r) * 512 + n] = v;
    outb[(size_t)(mb + r) * 512 + n] = f2bf(v);
  }
}

__global__ void k_style(const float* __restrict__ w, const float* __restrict__ sW,
                        const float* __restrict__ sb, float* __restrict__ s, int Cin) {
  int idx = blockIdx.x * blockDim.x + threadIdx.x;
  if (idx >= 16 * Cin) return;
  int b = idx / Cin, i = idx % Cin;
  const float* wr = w + (size_t)b * 512;
  const float* Wr = sW + (size_t)i * 512;
  float acc = 0.f;
#pragma unroll 4
  for (int k = 0; k < 512; ++k) acc += wr[k] * Wr[k];
  s[idx] = acc * 0.04419417382f + sb[i];
}

__global__ void k_wprep(const float* __restrict__ Wsrc, unsigned short* __restrict__ Wp,
                        float* __restrict__ Q, int Cout, int Cin, int KK, float scale) {
  int idx = blockIdx.x * blockDim.x + threadIdx.x;
  if (idx >= Cout * Cin) return;
  int o = idx / Cin, i = idx % Cin;
  float q = 0.f;
  for (int t = 0; t < KK; ++t) {
    float wv = Wsrc[(size_t)idx * KK + t];
    q += wv * wv;
    Wp[((size_t)t * Cout + o) * Cin + i] = f2bf(wv * scale);
  }
  Q[idx] = q;
}

__global__ void k_sigma(const float* __restrict__ s, const float* __restrict__ Q,
                        float* __restrict__ sig, int Cin, int Cout, float scale2) {
  int idx = blockIdx.x * blockDim.x + threadIdx.x;
  if (idx >= 16 * Cout) return;
  int b = idx / Cout, o = idx % Cout;
  const float* sr = s + (size_t)b * Cin;
  const float* qr = Q + (size_t)o * Cin;
  float acc = 0.f;
#pragma unroll 4
  for (int i = 0; i < Cin; ++i) acc += sr[i] * sr[i] * qr[i];
  sig[idx] = rsqrtf(acc * scale2 + 1e-6f);
}

__global__ void k_mod(const float* __restrict__ x, const float* __restrict__ s,
                      unsigned short* __restrict__ xm, int C, int H, int W, int total) {
  int idx = blockIdx.x * blockDim.x + threadIdx.x;
  if (idx >= total) return;
  int w = idx % W; int t = idx / W;
  int h = t % H; t /= H;
  int c = t % C; int b = t / C;
  float v = x[idx] * s[(size_t)b * C + c];
  xm[(((size_t)b * H + h) * W + w) * C + c] = f2bf(v);
}

// modulated 3x3 conv: implicit GEMM with double-buffered LDS weight tiles.
// Tap 0 staged via async-to-LDS; taps 1..8 prefetched by TDM while computing.
template <int CIN, int WPB>
__global__ __launch_bounds__(WPB * 32) void k_conv3t(
    const unsigned short* __restrict__ xm,  // [B][H][W][CIN] bf16
    const unsigned short* __restrict__ Wp,  // [9][Cout][CIN] bf16 (scaled)
    const float* __restrict__ sigma,        // [B][Cout]
    const float* __restrict__ nw,           // [Cout]
    const float* __restrict__ bias,         // [Cout]
    float* __restrict__ out,                // [B][Cout][H][W]
    int Cout, int H, int W, int nz) {
  __shared__ alignas(16) unsigned short Atile[2][16 * CIN];
  int npix = H * W;
  int ptiles = npix >> 4;
  int otiles = Cout >> 4;
  int pgroups = ptiles / WPB;
  int bid = blockIdx.x;
  int pg = bid % pgroups; bid /= pgroups;
  int ot = bid % otiles;  int b = bid / otiles;
  int wave = (int)(threadIdx.x >> 5);
  int lane = threadIdx.x & 31;
  int li = lane & 15;
  int pix = (pg * WPB + wave) * 16 + li;
  int ph = pix / W, pw = pix % W;

  v8f acc;
#pragma unroll
  for (int r = 0; r < 8; ++r) acc[r] = 0.f;

  constexpr int N4 = (16 * CIN) / 8;  // uint4 count of one tap tile
  auto stage = [&](const unsigned short* src, unsigned short* dstLds) {
#if HAVE_ASYNC
    for (int i = threadIdx.x; i < N4; i += WPB * 32)
      __builtin_amdgcn_global_load_async_to_lds_b128(
          TO_GLOBAL(src + 8 * i), TO_LDS(dstLds + 8 * i), 0, 0);
#else
    for (int i = threadIdx.x; i < N4; i += WPB * 32)
      ((uint4*)dstLds)[i] = ((const uint4*)src)[i];
#endif
  };

  // preload tap 0
  stage(Wp + (size_t)(ot * 16) * CIN, &Atile[0][0]);
#if HAVE_ASYNC
  __builtin_amdgcn_s_wait_asynccnt(0);
#endif
  __syncthreads();

  for (int t = 0; t < 9; ++t) {
    if (t < 8) {
      const unsigned short* nsrc = Wp + ((size_t)(t + 1) * Cout + ot * 16) * CIN;
#if HAVE_TDM
      if (threadIdx.x == 0)
        tdm_load_1d(nsrc, &Atile[(t + 1) & 1][0], 16 * CIN);
#else
      stage(nsrc, &Atile[(t + 1) & 1][0]);
#endif
    }
    int dy = t / 3 - 1, dx = t % 3 - 1;
    int ih = ph + dy, iw = pw + dx;
    bool ok = (ih >= 0) && (ih < H) && (iw >= 0) && (iw < W);
    const unsigned short* bptr = xm + (((size_t)b * H + ih) * W + iw) * CIN;
    const unsigned short* arow = &Atile[t & 1][li * CIN];
#pragma unroll
    for (int k0 = 0; k0 < CIN; k0 += 32) {
      v16bf A = loadA_vec(arow + k0);      // ds_load_b128 x2
      v16bf B = loadB_vec(bptr + k0, ok);  // global_load_b128 x2
      acc = wmma_bf16(A, B, acc);
    }
    if (t < 8) {
#if HAVE_TDM
      __builtin_amdgcn_s_wait_tensorcnt(0);
#elif HAVE_ASYNC
      __builtin_amdgcn_s_wait_asynccnt(0);
#endif
      __syncthreads();
    }
  }
  float nzv = gnoise((unsigned)b, (unsigned)ph, (unsigned)pw, (unsigned)nz);
  int mb = (lane & 16) ? 8 : 0;
#pragma unroll
  for (int r = 0; r < 8; ++r) {
    int o = ot * 16 + mb + r;
    float v = acc[r] * sigma[(size_t)b * Cout + o] + nw[o] * nzv + bias[o];
    out[(((size_t)b * Cout + o) * H + ph) * W + pw] = lrelu(v);
  }
}

// to_rgb: 1x1 modulated+demodulated conv to 3ch (+ optional upsampled rgb add)
__global__ void k_rgb(const unsigned short* __restrict__ xm,  // [B][H][W][C]
                      const unsigned short* __restrict__ Wp,  // [3][C]
                      const float* __restrict__ sigma,        // [B][3]
                      const float* __restrict__ bias,         // [3]
                      const float* __restrict__ up,           // [B][3][H][W] or null
                      float* __restrict__ out, int C, int H, int W) {
  int idx = blockIdx.x * blockDim.x + threadIdx.x;
  int npix = H * W;
  if (idx >= 16 * npix) return;
  int b = idx / npix;
  const unsigned short* p = xm + (size_t)idx * C;
  float a0 = 0.f, a1 = 0.f, a2 = 0.f;
  for (int i = 0; i < C; i += 8) {
    uint4 xv = *(const uint4*)(p + i);
    uint4 w0 = *(const uint4*)(Wp + i);
    uint4 w1 = *(const uint4*)(Wp + C + i);
    uint4 w2 = *(const uint4*)(Wp + 2 * C + i);
    unsigned int xs[4] = {xv.x, xv.y, xv.z, xv.w};
    unsigned int w0s[4] = {w0.x, w0.y, w0.z, w0.w};
    unsigned int w1s[4] = {w1.x, w1.y, w1.z, w1.w};
    unsigned int w2s[4] = {w2.x, w2.y, w2.z, w2.w};
#pragma unroll
    for (int j = 0; j < 4; ++j) {
      float xl = bflo(xs[j]), xh = bfhi(xs[j]);
      a0 += xl * bflo(w0s[j]) + xh * bfhi(w0s[j]);
      a1 += xl * bflo(w1s[j]) + xh * bfhi(w1s[j]);
      a2 += xl * bflo(w2s[j]) + xh * bfhi(w2s[j]);
    }
  }
  float a[3] = {a0, a1, a2};
#pragma unroll
  for (int c = 0; c < 3; ++c) {
    float v = lrelu(a[c] * sigma[(size_t)b * 3 + c] + bias[c]);
    size_t o = ((size_t)(b * 3 + c)) * npix + (idx % npix);
    out[o] = up ? up[o] + v : v;
  }
}

static __device__ inline int refl(int i, int n) {
  return (i < 0) ? -i : ((i >= n) ? 2 * n - 2 - i : i);
}

__global__ void k_up(const float* __restrict__ x, float* __restrict__ y,
                     int C, int H, int W, int total) {
  int idx = blockIdx.x * blockDim.x + threadIdx.x;
  if (idx >= total) return;
  int H2 = 2 * H, W2 = 2 * W;
  int w = idx % W2; int t = idx / W2;
  int h = t % H2; t /= H2;
  int c = t % C; int b = t / C;
  const float kk[3] = {0.25f, 0.5f, 0.25f};
  const float* xp = x + ((size_t)(b * C + c)) * H * W;
  float acc = 0.f;
#pragma unroll
  for (int dy = 0; dy < 3; ++dy) {
    int ih = refl(h + dy - 1, H2) >> 1;
#pragma unroll
    for (int dx = 0; dx < 3; ++dx) {
      int iw = refl(w + dx - 1, W2) >> 1;
      acc += kk[dy] * kk[dx] * xp[(size_t)ih * W + iw];
    }
  }
  y[idx] = acc;
}

__global__ void k_bcast(const float* __restrict__ cst, float* __restrict__ x,
                        int per, int total) {
  int idx = blockIdx.x * blockDim.x + threadIdx.x;
  if (idx >= total) return;
  x[idx] = cst[idx % per];
}

// ================= host =================

extern "C" void kernel_launch(void* const* d_in, const int* in_sizes, int n_in,
                              void* d_out, int out_size, void* d_ws, size_t ws_size,
                              hipStream_t stream) {
  (void)in_sizes; (void)n_in; (void)out_size; (void)ws_size;
  auto in = [&](int i) { return (const float*)d_in[i]; };
  char* ws = (char*)d_ws;
  size_t off = 0;
  auto alloc = [&](size_t bytes) {
    void* p = (void*)(ws + off);
    off += (bytes + 255) & ~(size_t)255;
    return p;
  };
  float* wF   = (float*)alloc((size_t)16 * 512 * 4);           // f32 latent (for styles)
  unsigned short* wbfA = (unsigned short*)alloc((size_t)16 * 512 * 2);
  unsigned short* wbfB = (unsigned short*)alloc((size_t)16 * 512 * 2);
  unsigned short* Wbf  = (unsigned short*)alloc((size_t)512 * 512 * 2);
  float* sbuf = (float*)alloc((size_t)16 * 512 * 4);
  float* sig  = (float*)alloc((size_t)16 * 512 * 4);
  float* Q    = (float*)alloc((size_t)512 * 512 * 4);
  unsigned short* Wp = (unsigned short*)alloc((size_t)9 * 512 * 512 * 2);
  const size_t XMAX = (size_t)16 * 64 * 128 * 128;
  float* xA = (float*)alloc(XMAX * 4);
  float* xB = (float*)alloc(XMAX * 4);
  unsigned short* xm = (unsigned short*)alloc(XMAX * 2);
  float* rgbA = (float*)alloc((size_t)16 * 3 * 128 * 128 * 4);
  float* rgbB = (float*)alloc((size_t)16 * 3 * 128 * 128 * 4);

  auto cdiv = [](long long a, int b) { return (int)((a + b - 1) / b); };

  // ---- mapping network: pre-convert weights/activations to bf16, WMMA GEMMs ----
  k_f2bf<<<cdiv(16 * 512 / 8, 256), 256, 0, stream>>>(in(0), wbfA, 16 * 512 / 8);
  for (int l = 0; l < 8; ++l) {
    k_f2bf<<<cdiv(512 * 512 / 8, 256), 256, 0, stream>>>(in(1 + 2 * l), Wbf,
                                                         512 * 512 / 8);
    const unsigned short* src = (l & 1) ? wbfB : wbfA;
    unsigned short* dst = (l & 1) ? wbfA : wbfB;
    k_map2<<<32, 32, 0, stream>>>(src, Wbf, in(2 + 2 * l), wF, dst);
  }
  const float* wlat = wF;

  auto launch_conv = [&](int Cin, int Cout, int H, int W, const float* nw,
                         const float* bi, float* x_out, int nz) {
    int npix = H * W, ptiles = npix / 16, otiles = Cout / 16;
    if (npix < 64) {  // 4x4 layer: Cin==512, single-wave blocks
      k_conv3t<512, 1><<<16 * otiles * ptiles, 32, 0, stream>>>(
          xm, Wp, sig, nw, bi, x_out, Cout, H, W, nz);
      return;
    }
    int nb = 16 * otiles * (ptiles / 4);
    switch (Cin) {
      case 512: k_conv3t<512, 4><<<nb, 128, 0, stream>>>(xm, Wp, sig, nw, bi, x_out, Cout, H, W, nz); break;
      case 256: k_conv3t<256, 4><<<nb, 128, 0, stream>>>(xm, Wp, sig, nw, bi, x_out, Cout, H, W, nz); break;
      case 128: k_conv3t<128, 4><<<nb, 128, 0, stream>>>(xm, Wp, sig, nw, bi, x_out, Cout, H, W, nz); break;
      case 64:  k_conv3t<64, 4><<<nb, 128, 0, stream>>>(xm, Wp, sig, nw, bi, x_out, Cout, H, W, nz); break;
      default:  k_conv3t<32, 4><<<nb, 128, 0, stream>>>(xm, Wp, sig, nw, bi, x_out, Cout, H, W, nz); break;
    }
  };

  auto run_layer = [&](const float* x_in, float* x_out, int Cin, int Cout, int H, int W,
                       const float* sW, const float* sb, const float* cW,
                       const float* nw, const float* bi, int nz) {
    long long npx = (long long)H * W;
    k_style<<<cdiv(16LL * Cin, 256), 256, 0, stream>>>(wlat, sW, sb, sbuf, Cin);
    k_wprep<<<cdiv((long long)Cout * Cin, 256), 256, 0, stream>>>(
        cW, Wp, Q, Cout, Cin, 9, 1.f / sqrtf((float)Cin * 9.f));
    k_sigma<<<cdiv(16LL * Cout, 256), 256, 0, stream>>>(sbuf, Q, sig, Cin, Cout,
                                                        1.f / ((float)Cin * 9.f));
    int tmod = (int)(16 * Cin * npx);
    k_mod<<<cdiv(tmod, 256), 256, 0, stream>>>(x_in, sbuf, xm, Cin, H, W, tmod);
    launch_conv(Cin, Cout, H, W, nw, bi, x_out, nz);
  };

  auto run_rgb = [&](const float* x_in, int C, int H, int W, const float* sW,
                     const float* sb, const float* cW, const float* bi,
                     const float* up, float* outp) {
    long long npx = (long long)H * W;
    k_style<<<cdiv(16LL * C, 256), 256, 0, stream>>>(wlat, sW, sb, sbuf, C);
    k_wprep<<<cdiv(3LL * C, 256), 256, 0, stream>>>(cW, Wp, Q, 3, C, 1,
                                                    1.f / sqrtf((float)C));
    k_sigma<<<1, 256, 0, stream>>>(sbuf, Q, sig, C, 3, 1.f / (float)C);
    int tmod = (int)(16 * C * npx);
    k_mod<<<cdiv(tmod, 256), 256, 0, stream>>>(x_in, sbuf, xm, C, H, W, tmod);
    k_rgb<<<cdiv(16 * npx, 256), 256, 0, stream>>>(xm, Wp, sig, bi, up, outp, C, H, W);
  };

  int feats[6] = {512, 512, 256, 128, 64, 32};

  {
    int total = 16 * 512 * 16;
    k_bcast<<<cdiv(total, 256), 256, 0, stream>>>(in(17), xA, 512 * 16, total);
  }
  run_layer(xA, xB, 512, 512, 4, 4, in(18), in(19), in(20), in(21), in(22), 0);
  run_rgb(xB, 512, 4, 4, in(23), in(24), in(25), in(26), nullptr, rgbA);

  float* x_cur = xB;
  float* x_oth = xA;
  int nz = 1;
  for (int blk = 0; blk < 5; ++blk) {
    int Cin = feats[blk], Cout = feats[blk + 1];
    int Hin = 4 << blk, Hout = 8 << blk;
    int base = 27 + 14 * blk;
    int tup = 16 * Cin * Hout * Hout;
    k_up<<<cdiv(tup, 256), 256, 0, stream>>>(x_cur, x_oth, Cin, Hin, Hin, tup);
    run_layer(x_oth, x_cur, Cin, Cout, Hout, Hout, in(base + 0), in(base + 1),
              in(base + 2), in(base + 3), in(base + 4), nz++);
    run_layer(x_cur, x_oth, Cout, Cout, Hout, Hout, in(base + 5), in(base + 6),
              in(base + 7), in(base + 8), in(base + 9), nz++);
    int tur = 16 * 3 * Hout * Hout;
    k_up<<<cdiv(tur, 256), 256, 0, stream>>>(rgbA, rgbB, 3, Hin, Hin, tur);
    float* rdst = (blk == 4) ? (float*)d_out : rgbA;
    run_rgb(x_oth, Cout, Hout, Hout, in(base + 10), in(base + 11), in(base + 12),
            in(base + 13), rgbB, rdst);
    float* t = x_cur; x_cur = x_oth; x_oth = t;
  }
}
